// PMETorchPME_46969762349278
// MI455X (gfx1250) — matrix operand order repaired and checked by reading, not compiled, not run on hardware
//
#include <hip/hip_runtime.h>
#include <hip/hip_bf16.h>
#include <math.h>

#define NSIDE 120
#define NROWS (NSIDE*NSIDE)          /* 14400 lines per axis transform */
#define N3    (NSIDE*NSIDE*NSIDE)    /* 1,728,000 mesh points */
#define TSIDE 128                    /* padded table side */
#define TABP  (TSIDE*TSIDE)          /* one padded 128x128 twiddle table */
#define KPAD  128                    /* K padded to 4 chunks of 32 */
#define PI_F  3.14159265358979323846f

typedef __bf16 bf16_t;
typedef __attribute__((ext_vector_type(16))) __bf16 v16bf;
typedef __attribute__((ext_vector_type(8)))  __bf16 v8bf;
typedef __attribute__((ext_vector_type(8)))  float  v8f;
typedef unsigned int u32x4 __attribute__((ext_vector_type(4)));
typedef int          i32x4 __attribute__((ext_vector_type(4)));
typedef int          i32x8 __attribute__((ext_vector_type(8)));

#define WMMA_BF16(A,B,C) \
  __builtin_amdgcn_wmma_f32_16x16x32_bf16(false,(A),false,(B),(short)0,(C),false,false)

#if __has_builtin(__builtin_amdgcn_tensor_load_to_lds) && __has_builtin(__builtin_amdgcn_s_wait_tensorcnt)
#define USE_TDM 1
#else
#define USE_TDM 0
#endif

// ---------------------------------------------------------------- helpers
__device__ __forceinline__ void inv3x3(const float* b, float* inv) {
  float a00=b[0],a01=b[1],a02=b[2],a10=b[3],a11=b[4],a12=b[5],a20=b[6],a21=b[7],a22=b[8];
  float det = a00*(a11*a22-a12*a21) - a01*(a10*a22-a12*a20) + a02*(a10*a21-a11*a20);
  float id = 1.0f/det;
  inv[0]=(a11*a22-a12*a21)*id; inv[1]=(a02*a21-a01*a22)*id; inv[2]=(a01*a12-a02*a11)*id;
  inv[3]=(a12*a20-a10*a22)*id; inv[4]=(a00*a22-a02*a20)*id; inv[5]=(a02*a10-a00*a12)*id;
  inv[6]=(a10*a21-a11*a20)*id; inv[7]=(a01*a20-a00*a21)*id; inv[8]=(a00*a11-a01*a10)*id;
}

// Lagrange ORDER=6 interpolation weights; x in [-0.5,0.5), nodes t_k = k-2.5
__device__ __forceinline__ void lag_weights6(float x, float* w) {
  const float invden[6] = {-1.f/120.f, 1.f/24.f, -1.f/12.f, 1.f/12.f, -1.f/24.f, 1.f/120.f};
  #pragma unroll
  for (int j = 0; j < 6; ++j) {
    float p = 1.0f;
    #pragma unroll
    for (int k = 0; k < 6; ++k)
      if (k != j) p *= (x - ((float)k - 2.5f));
    w[j] = p * invden[j];
  }
}

__device__ __forceinline__ void splitbf(float x, bf16_t& h, bf16_t& l) {
  h = (bf16_t)x;
  l = (bf16_t)(x - (float)h);
}

// Issue a TDM 2D tile load: tile_dim0=120 f32 elems x tile_dim1=16 rows,
// row stride 120 elems, from `gsrc` into LDS byte-address `lds_addr`.
__device__ __forceinline__ void tdm_load_rowblock(const float* gsrc, unsigned lds_addr) {
#if USE_TDM
  unsigned long long ga = (unsigned long long)(const void*)gsrc;
  u32x4 g0;
  g0[0] = 1u;                                   // count=1 valid descriptor
  g0[1] = lds_addr;                             // lds_addr (bytes)
  g0[2] = (unsigned)(ga & 0xFFFFFFFFu);         // global_addr[31:0]
  g0[3] = (unsigned)((ga >> 32) & 0x1FFFFFFu) | (2u << 30);  // addr[56:32] | type=2
  i32x8 g1;
  g1[0] = (int)(2u << 16);                      // data_size=2 (4 bytes)
  g1[1] = (int)(120u << 16);                    // tensor_dim0[15:0] = 120 (bits 63:48)
  g1[2] = (int)(16u  << 16);                    // tensor_dim1[15:0] = 16  (bits 95:80)
  g1[3] = (int)(120u << 16);                    // tile_dim0 = 120 (bits 127:112)
  g1[4] = 16;                                   // tile_dim1 = 16 (bits 143:128)
  g1[5] = 120;                                  // tensor_dim0_stride[31:0] = 120
  g1[6] = 0;                                    // stride hi / dim1_stride lo
  g1[7] = 0;
  i32x4 gz = {0, 0, 0, 0};
#if defined(__clang_major__) && __clang_major__ >= 23
  i32x8 gz8 = {0, 0, 0, 0, 0, 0, 0, 0};
  __builtin_amdgcn_tensor_load_to_lds(g0, g1, gz, gz, gz8, 0);
#else
  __builtin_amdgcn_tensor_load_to_lds(g0, g1, gz, gz, 0);
#endif
#else
  (void)gsrc; (void)lds_addr;
#endif
}

// ---------------------------------------------------------------- kernels
__global__ void pme_zero(float* a_re, float* accs, int n) {
  int i = blockIdx.x * blockDim.x + threadIdx.x;
  if (i < n) a_re[i] = 0.0f;
  if (i < 8) accs[i] = 0.0f;
}

// Twiddle tables, column-major + zero-padded to 128x128: tab[n*128 + z].
// [0..3]=fwd (cos_hi,cos_lo,msin_hi,msin_lo)  [4..7]=inv (cos_hi,cos_lo,sin_hi,sin_lo)
__global__ void pme_tables(bf16_t* tabs) {
  int i = blockIdx.x * blockDim.x + threadIdx.x;
  if (i >= TABP) return;
  int nn = i / TSIDE, z = i % TSIDE;
  float c = 0.0f, s = 0.0f;
  if (nn < NSIDE && z < NSIDE) {
    int phase = (z * nn) % NSIDE;               // exact integer phase reduction
    float th = 2.0f * PI_F * (float)phase / (float)NSIDE;
    __sincosf(th, &s, &c);
  }
  bf16_t h, l;
  splitbf(c,  h, l); tabs[0*TABP+i]=h; tabs[1*TABP+i]=l;   // fwd cos
  splitbf(-s, h, l); tabs[2*TABP+i]=h; tabs[3*TABP+i]=l;   // fwd -sin (e^{-i t})
  splitbf(c,  h, l); tabs[4*TABP+i]=h; tabs[5*TABP+i]=l;   // inv cos
  splitbf(s,  h, l); tabs[6*TABP+i]=h; tabs[7*TABP+i]=l;   // inv +sin (e^{+i t})
}

__global__ void pme_spread(const float* __restrict__ coords, const float* __restrict__ box,
                           const float* __restrict__ charges, float* __restrict__ mesh, int n) {
  int i = blockIdx.x * blockDim.x + threadIdx.x;
  if (i >= n) return;
  float b[9], inv[9];
  #pragma unroll
  for (int t = 0; t < 9; ++t) b[t] = box[t];
  inv3x3(b, inv);
  float c0 = coords[3*i+0], c1 = coords[3*i+1], c2 = coords[3*i+2];
  float w[3][6]; int g[3][6];
  #pragma unroll
  for (int d = 0; d < 3; ++d) {
    float frac = c0*inv[0*3+d] + c1*inv[1*3+d] + c2*inv[2*3+d];
    float pos  = frac * (float)NSIDE;
    float fl   = floorf(pos);
    lag_weights6(pos - (fl + 0.5f), w[d]);
    int i0 = (int)fl;
    #pragma unroll
    for (int o = 0; o < 6; ++o) {
      int gg = (i0 + o - 2) % NSIDE; if (gg < 0) gg += NSIDE;
      g[d][o] = gg;
    }
  }
  float q = charges[i];
  for (int a = 0; a < 6; ++a)
    for (int bb = 0; bb < 6; ++bb) {
      float wab = w[0][a] * w[1][bb] * q;
      int base = (g[0][a]*NSIDE + g[1][bb]) * NSIDE;
      for (int cc = 0; cc < 6; ++cc)
        atomicAdd(&mesh[base + g[2][cc]], wab * w[2][cc]);
    }
}

// One 256-thread block (8 waves) per 16-row block. TDM stages the f32 row block
// into LDS; converted once to split-bf16 (hi+lo) in LDS; each wave computes one
// 16x16 column tile of OUT[R,120] = IN[R,120] * (C + iS) with
// v_wmma_f32_16x16x32_bf16 (hi*hi + hi*lo + lo*hi). A fragments: 2x ds_load_b128
// per fragment; B fragments: one aligned 32B global load from the padded
// column-major tables. Cyclic permute out[c][a][b] = D[(a,b)][c] fused in store.
template<int HAS_IM>
__global__ __launch_bounds__(256) void pme_dft_gemm(
    const float* __restrict__ in_re, const float* __restrict__ in_im,
    float* __restrict__ out_re, float* __restrict__ out_im,
    const bf16_t* __restrict__ c_hi, const bf16_t* __restrict__ c_lo,
    const bf16_t* __restrict__ s_hi, const bf16_t* __restrict__ s_lo) {
  __shared__ __align__(16) float  bufRe[16*NSIDE];
  __shared__ __align__(16) float  bufIm[16*NSIDE];
  __shared__ __align__(16) bf16_t arHi[16*KPAD], arLo[16*KPAD];
  __shared__ __align__(16) bf16_t aiHi[16*KPAD], aiLo[16*KPAD];

  const int tid  = threadIdx.x;
  const int lane = tid & 31;
  const int wave = tid >> 5;           // 8 waves -> 8 column tiles
  const int m    = lane & 15;
  const int half = lane >> 4;
  const int row0 = blockIdx.x * 16;
  const int n    = wave * 16 + (lane & 15);

  // ---- stage the 16x120 f32 row block(s) into LDS --------------------------
#if USE_TDM
  if (tid < 32) {                      // wave 0 issues the DMA(s)
    tdm_load_rowblock(in_re + (size_t)row0 * NSIDE, (unsigned)(size_t)&bufRe[0]);
    if (HAS_IM)
      tdm_load_rowblock(in_im + (size_t)row0 * NSIDE, (unsigned)(size_t)&bufIm[0]);
    __builtin_amdgcn_s_wait_tensorcnt(0);
  }
#else
  for (int idx = tid; idx < 16 * NSIDE; idx += 256) {
    bufRe[idx] = in_re[(size_t)row0 * NSIDE + idx];
    if (HAS_IM) bufIm[idx] = in_im[(size_t)row0 * NSIDE + idx];
  }
#endif
  __syncthreads();

  // ---- convert once to split-bf16, K padded 120->128 with zeros ------------
  for (int idx = tid; idx < 16 * KPAD; idx += 256) {
    int r = idx >> 7, z = idx & (KPAD - 1);
    float xr = (z < NSIDE) ? bufRe[r * NSIDE + z] : 0.0f;
    bf16_t h, l; splitbf(xr, h, l);
    arHi[idx] = h; arLo[idx] = l;
    if (HAS_IM) {
      float xi = (z < NSIDE) ? bufIm[r * NSIDE + z] : 0.0f;
      splitbf(xi, h, l);
      aiHi[idx] = h; aiLo[idx] = l;
    }
  }
  __syncthreads();

  v8f accRR = {}, accRI = {}, accIR = {}, accII = {};

  const size_t tcol = (size_t)n * TSIDE;   // column base in padded tables

  for (int kb = 0; kb < NSIDE; kb += 32) {
    // A fragment: elements e -> K = kb+16*half+e (contiguous)
    const int abase = m * KPAD + kb + (half << 4);
    const v8bf* pr_hi = (const v8bf*)&arHi[abase];
    const v8bf* pr_lo = (const v8bf*)&arLo[abase];
    v16bf ar_hi = __builtin_shufflevector(pr_hi[0], pr_hi[1], 0,1,2,3,4,5,6,7,8,9,10,11,12,13,14,15);
    v16bf ar_lo = __builtin_shufflevector(pr_lo[0], pr_lo[1], 0,1,2,3,4,5,6,7,8,9,10,11,12,13,14,15);

    // B fragment: elements e -> z = kb + 16*half + e (contiguous, padded tables)
    const size_t bbase = tcol + kb + (half << 4);
    v16bf bc_hi = *(const v16bf*)(c_hi + bbase);
    v16bf bc_lo = *(const v16bf*)(c_lo + bbase);
    v16bf bs_hi = *(const v16bf*)(s_hi + bbase);
    v16bf bs_lo = *(const v16bf*)(s_lo + bbase);

    accRR = WMMA_BF16(ar_hi, bc_hi, accRR);
    accRR = WMMA_BF16(ar_hi, bc_lo, accRR);
    accRR = WMMA_BF16(ar_lo, bc_hi, accRR);
    accRI = WMMA_BF16(ar_hi, bs_hi, accRI);
    accRI = WMMA_BF16(ar_hi, bs_lo, accRI);
    accRI = WMMA_BF16(ar_lo, bs_hi, accRI);
    if (HAS_IM) {
      const v8bf* pi_hi = (const v8bf*)&aiHi[abase];
      const v8bf* pi_lo = (const v8bf*)&aiLo[abase];
      v16bf ai_hi = __builtin_shufflevector(pi_hi[0], pi_hi[1], 0,1,2,3,4,5,6,7,8,9,10,11,12,13,14,15);
      v16bf ai_lo = __builtin_shufflevector(pi_lo[0], pi_lo[1], 0,1,2,3,4,5,6,7,8,9,10,11,12,13,14,15);
      accIR = WMMA_BF16(ai_hi, bc_hi, accIR);
      accIR = WMMA_BF16(ai_hi, bc_lo, accIR);
      accIR = WMMA_BF16(ai_lo, bc_hi, accIR);
      accII = WMMA_BF16(ai_hi, bs_hi, accII);
      accII = WMMA_BF16(ai_hi, bs_lo, accII);
      accII = WMMA_BF16(ai_lo, bs_hi, accII);
    }
  }

  // D layout: VGPR r -> M = r + 8*half, N = lane%16.
  // Fused cyclic permute: row=(a*120+b), store at ((n*120 + a)*120 + b).
  if (n < NSIDE) {
    #pragma unroll
    for (int r = 0; r < 8; ++r) {
      int row = row0 + r + half * 8;
      int a = row / NSIDE, b = row - a * NSIDE;
      size_t o = (size_t)n * NROWS + (size_t)a * NSIDE + b;
      out_re[o] = accRR[r] - accII[r];
      out_im[o] = accRI[r] + accIR[r];
    }
  }
}

__global__ void pme_gmult(float* __restrict__ re, float* __restrict__ im,
                          const float* __restrict__ box) {
  int i = blockIdx.x * blockDim.x + threadIdx.x;
  if (i >= N3) return;
  float b[9], inv[9];
  #pragma unroll
  for (int t = 0; t < 9; ++t) b[t] = box[t];
  inv3x3(b, inv);
  int i0 = i / NROWS;
  int rem = i - i0 * NROWS;
  int i1 = rem / NSIDE, i2 = rem % NSIDE;
  float m0 = (float)(i0 < NSIDE/2 ? i0 : i0 - NSIDE);
  float m1 = (float)(i1 < NSIDE/2 ? i1 : i1 - NSIDE);
  float m2 = (float)(i2 < NSIDE/2 ? i2 : i2 - NSIDE);
  float ksq = 0.0f;
  #pragma unroll
  for (int d = 0; d < 3; ++d) {
    float kd = 2.0f * PI_F * (m0*inv[0*3+d] + m1*inv[1*3+d] + m2*inv[2*3+d]);
    ksq += kd * kd;
  }
  float G = (ksq > 0.0f) ? 4.0f * PI_F * __expf(-0.5f * ksq) / ksq : 0.0f; // ALPHA=1
  re[i] *= G;
  im[i] *= G;
}

__global__ void pme_gather(const float* __restrict__ coords, const float* __restrict__ box,
                           const float* __restrict__ charges, const float* __restrict__ mesh,
                           float* __restrict__ accs, int n) {
  __shared__ float s0[256], s1[256], s2[256];
  int tid = threadIdx.x;
  int i = blockIdx.x * blockDim.x + tid;
  float e0 = 0.0f, e1 = 0.0f, e2 = 0.0f;
  if (i < n) {
    float b[9], inv[9];
    #pragma unroll
    for (int t = 0; t < 9; ++t) b[t] = box[t];
    inv3x3(b, inv);
    float c0 = coords[3*i+0], c1 = coords[3*i+1], c2 = coords[3*i+2];
    float w[3][6]; int g[3][6];
    #pragma unroll
    for (int d = 0; d < 3; ++d) {
      float frac = c0*inv[0*3+d] + c1*inv[1*3+d] + c2*inv[2*3+d];
      float pos  = frac * (float)NSIDE;
      float fl   = floorf(pos);
      lag_weights6(pos - (fl + 0.5f), w[d]);
      int i0 = (int)fl;
      #pragma unroll
      for (int o = 0; o < 6; ++o) {
        int gg = (i0 + o - 2) % NSIDE; if (gg < 0) gg += NSIDE;
        g[d][o] = gg;
      }
    }
    float pot = 0.0f;
    for (int a = 0; a < 6; ++a)
      for (int bb = 0; bb < 6; ++bb) {
        float wab = w[0][a] * w[1][bb];
        int base = (g[0][a]*NSIDE + g[1][bb]) * NSIDE;
        float acc = 0.0f;
        for (int cc = 0; cc < 6; ++cc)
          acc += w[2][cc] * mesh[base + g[2][cc]];
        pot += wab * acc;
      }
    float q = charges[i];
    e0 = q * pot; e1 = q; e2 = q * q;
  }
  s0[tid] = e0; s1[tid] = e1; s2[tid] = e2;
  __syncthreads();
  for (int s = 128; s > 0; s >>= 1) {
    if (tid < s) { s0[tid]+=s0[tid+s]; s1[tid]+=s1[tid+s]; s2[tid]+=s2[tid+s]; }
    __syncthreads();
  }
  if (tid == 0) {
    atomicAdd(&accs[0], s0[0]);
    atomicAdd(&accs[1], s1[0]);
    atomicAdd(&accs[2], s2[0]);
  }
}

__global__ void pme_final(const float* __restrict__ accs, const float* __restrict__ box,
                          float* __restrict__ out) {
  if (threadIdx.x != 0 || blockIdx.x != 0) return;
  float a00=box[0],a01=box[1],a02=box[2],a10=box[3],a11=box[4],a12=box[5],a20=box[6],a21=box[7],a22=box[8];
  float det = a00*(a11*a22-a12*a21) - a01*(a10*a22-a12*a20) + a02*(a10*a21-a11*a20);
  float V = fabsf(det);
  float qg = accs[0], qs = accs[1], q2 = accs[2];
  // E = 0.5 * ( sum(q*gather)/V - sqrt(2/pi)/alpha * sum(q^2) - 2*pi*alpha^2*Qtot^2/V )
  float E = 0.5f * (qg / V - sqrtf(2.0f / PI_F) * q2 - 2.0f * PI_F * qs * qs / V);
  out[0] = E;
}

// ---------------------------------------------------------------- launch
extern "C" void kernel_launch(void* const* d_in, const int* in_sizes, int n_in,
                              void* d_out, int out_size, void* d_ws, size_t ws_size,
                              hipStream_t stream) {
  (void)n_in; (void)out_size; (void)ws_size;
  const float* coords  = (const float*)d_in[0];
  const float* box     = (const float*)d_in[1];
  const float* charges = (const float*)d_in[2];
  const int natoms = in_sizes[0] / 3;

  float* ws   = (float*)d_ws;
  float* A_re = ws + (size_t)0 * N3;
  float* A_im = ws + (size_t)1 * N3;
  float* B_re = ws + (size_t)2 * N3;
  float* B_im = ws + (size_t)3 * N3;
  float* accs = ws + (size_t)4 * N3;
  bf16_t* tabs = (bf16_t*)(accs + 8);
  const bf16_t* fc_hi = tabs + (size_t)0*TABP; const bf16_t* fc_lo = tabs + (size_t)1*TABP;
  const bf16_t* fs_hi = tabs + (size_t)2*TABP; const bf16_t* fs_lo = tabs + (size_t)3*TABP;
  const bf16_t* ic_hi = tabs + (size_t)4*TABP; const bf16_t* ic_lo = tabs + (size_t)5*TABP;
  const bf16_t* is_hi = tabs + (size_t)6*TABP; const bf16_t* is_lo = tabs + (size_t)7*TABP;

  pme_zero<<<(N3 + 255) / 256, 256, 0, stream>>>(A_re, accs, N3);
  pme_tables<<<(TABP + 255) / 256, 256, 0, stream>>>(tabs);
  pme_spread<<<(natoms + 255) / 256, 256, 0, stream>>>(coords, box, charges, A_re, natoms);

  const int ggrid = NROWS / 16;   // 900 row blocks; 8 column tiles = 8 waves/block
  // forward: 3 axis transforms, output permuted in epilogue, ping-pong A<->B
  pme_dft_gemm<0><<<ggrid, 256, 0, stream>>>(A_re, A_im, B_re, B_im, fc_hi, fc_lo, fs_hi, fs_lo);
  pme_dft_gemm<1><<<ggrid, 256, 0, stream>>>(B_re, B_im, A_re, A_im, fc_hi, fc_lo, fs_hi, fs_lo);
  pme_dft_gemm<1><<<ggrid, 256, 0, stream>>>(A_re, A_im, B_re, B_im, fc_hi, fc_lo, fs_hi, fs_lo);
  // B now holds the spectrum in natural [kx][ky][kz] order
  pme_gmult<<<(N3 + 255) / 256, 256, 0, stream>>>(B_re, B_im, box);
  // inverse: 3 axis transforms
  pme_dft_gemm<1><<<ggrid, 256, 0, stream>>>(B_re, B_im, A_re, A_im, ic_hi, ic_lo, is_hi, is_lo);
  pme_dft_gemm<1><<<ggrid, 256, 0, stream>>>(A_re, A_im, B_re, B_im, ic_hi, ic_lo, is_hi, is_lo);
  pme_dft_gemm<1><<<ggrid, 256, 0, stream>>>(B_re, B_im, A_re, A_im, ic_hi, ic_lo, is_hi, is_lo);
  // A_re now holds pot_mesh in natural [x][y][z] order

  pme_gather<<<(natoms + 255) / 256, 256, 0, stream>>>(coords, box, charges, A_re, accs, natoms);
  pme_final<<<1, 1, 0, stream>>>(accs, box, (float*)d_out);
}
